// EUGATGNN_61899068670492
// MI455X (gfx1250) — matrix operand order repaired and checked by reading, not compiled, not run on hardware
//
#include <hip/hip_runtime.h>
#include <hip/hip_bf16.h>

// EGAT GNN (2 layers) for MI455X / gfx1250, wave32 + WMMA f32_16x16x32_f16.
// N=50000 nodes, E=800000 edges, D=128. Both N and E divisible by 16.

#define NNODES 50000
#define NEDGES 800000
#define DIM    128

typedef __attribute__((ext_vector_type(16))) _Float16 v16h;
typedef __attribute__((ext_vector_type(4)))  _Float16 v4h;
typedef __attribute__((ext_vector_type(8)))  float    v8f;

// ---------------------------------------------------------------------------
// WMMA fragment helpers (this code's contiguous K order, same for A and B; not AMD's table order)
// A (16x32, MxK): lane L<16 -> M=L, K runs [kb..kb+15], kb=0;
//                 lane L>=16 -> same M=L-16, kb=16.
// B (32x16, KxN): lane L -> N=L&15, VGPR i holds K = 16*(L>>4) + i (f16 pairs).
// C/D (16x16 f32): lane L -> N=L&15, acc[r] = row (r + 8*(L>>4)).
// ---------------------------------------------------------------------------
__device__ __forceinline__ v16h load_a_frag(const float* __restrict__ base,
                                            int row0, int k0, int lane) {
  const int m  = lane & 15;
  const int kb = k0 + ((lane >> 4) << 4);
  const float* p = base + (size_t)(row0 + m) * DIM + kb;
  v16h a;
#pragma unroll
  for (int i = 0; i < 8; ++i) a[i] = (_Float16)p[i];
#pragma unroll
  for (int i = 0; i < 8; ++i) a[8 + i] = (_Float16)p[8 + i];
  return a;
}

// B[k][n] = W[j0+n][k] for out = x @ W^T (W row-major [Dout][Din]) -> the
// per-lane 16 K-values are CONTIGUOUS: one 64B fp32 run, cvt to f16.
__device__ __forceinline__ v16h load_b_frag_g(const float* __restrict__ W,
                                              int j0, int k0, int lane) {
  const int n  = lane & 15;
  const int kb = k0 + ((lane >> 4) << 4);
  const float* p = W + (size_t)(j0 + n) * DIM + kb;
  v16h b;
#pragma unroll
  for (int i = 0; i < 16; ++i) b[i] = (_Float16)p[i];
  return b;
}

// Same fragment but from an LDS-resident f16 copy of W: 32B-aligned vector
// read -> two ds_load_b128, no conversions in the hot loop.
__device__ __forceinline__ v16h load_b_frag_lds(const _Float16* sW,
                                                int j0, int k0, int lane) {
  const int n  = lane & 15;
  const int kb = k0 + ((lane >> 4) << 4);
  return *(const v16h*)(sW + (j0 + n) * DIM + kb);
}

__device__ __forceinline__ v8f wmma_f16(v16h a, v16h b, v8f c) {
  return __builtin_amdgcn_wmma_f32_16x16x32_f16(
      /*neg_a=*/false, a, /*neg_b=*/false, b,
      /*c_mod=*/(short)0, c, /*reuse_a=*/false, /*reuse_b=*/false);
}

// monotonic float<->uint encoding so segment-max can use integer atomicMax
__device__ __forceinline__ unsigned enc_f32(float f) {
  unsigned u = __float_as_uint(f);
  return (u & 0x80000000u) ? ~u : (u | 0x80000000u);
}
__device__ __forceinline__ float dec_f32(unsigned u) {
  unsigned v = (u & 0x80000000u) ? (u & 0x7FFFFFFFu) : ~u;
  return __uint_as_float(v);
}

// ---------------------------------------------------------------------------
// Kernel 1: fused node GEMMs.  xWni = x@Wni^T, xWnj = x@Wnj^T, hproj = x@Wnode^T
// One wave per 16x16 tile; A fragment loaded once, reused across 3 weights.
// (No LDS staging: each block consumes each weight element exactly once.)
// ---------------------------------------------------------------------------
__global__ __launch_bounds__(256)
void egat_gemm3(const float* __restrict__ x,
                const float* __restrict__ W0, const float* __restrict__ W1,
                const float* __restrict__ W2,
                float* __restrict__ o0, float* __restrict__ o1,
                float* __restrict__ o2, int wtiles) {
  const int wave = (int)((blockIdx.x * blockDim.x + threadIdx.x) >> 5);
  if (wave >= wtiles) return;                   // wave-uniform: EXEC stays full
  const int lane  = threadIdx.x & 31;
  const int row0  = (wave >> 3) << 4;
  const int j0    = (wave & 7) << 4;
  if (lane == 0) {                              // global_prefetch_b8 touch
    __builtin_prefetch(W0, 0, 3);
    __builtin_prefetch(W1, 0, 3);
    __builtin_prefetch(W2, 0, 3);
  }
  v8f a0 = {}, a1 = {}, a2 = {};
#pragma unroll
  for (int kc = 0; kc < 4; ++kc) {
    v16h a = load_a_frag(x, row0, kc * 32, lane);
    a0 = wmma_f16(a, load_b_frag_g(W0, j0, kc * 32, lane), a0);
    a1 = wmma_f16(a, load_b_frag_g(W1, j0, kc * 32, lane), a1);
    a2 = wmma_f16(a, load_b_frag_g(W2, j0, kc * 32, lane), a2);
  }
  const int n = lane & 15, hi = lane >> 4;
#pragma unroll
  for (int r = 0; r < 8; ++r) {
    const size_t off = (size_t)(row0 + r + 8 * hi) * DIM + j0 + n;
    o0[off] = a0[r];
    o1[off] = a1[r];
    o2[off] = a2[r];
  }
}

// ---------------------------------------------------------------------------
// Kernel 2: fused edge GEMM + gather + leaky_relu + attn-score + segment-max.
// One wave per 16 edges. f = leaky(eW + xWni[src] + xWnj[dst]); score = f.attn
// W_fij staged once per block into LDS as f16 (32 KB of 320 KB/WGP): every
// wave reuses the full weight 8x, so this cuts L2 weight traffic 8x and
// removes all B-side f32->f16 conversions from the hot loop.
// ---------------------------------------------------------------------------
__global__ __launch_bounds__(256)
void egat_edge_fused(const float* __restrict__ ein,
                     const float* __restrict__ Wfij,
                     const float* __restrict__ xWni,
                     const float* __restrict__ xWnj,
                     const float* __restrict__ attn,
                     const int* __restrict__ src, const int* __restrict__ dst,
                     float* __restrict__ fout,           // may be nullptr
                     float* __restrict__ score,
                     unsigned* __restrict__ menc, int etiles) {
  __shared__ __align__(32) _Float16 sW[DIM * DIM];       // 32 KB, W[j][k] f16

  // cooperative stage + convert: 16384 floats / 256 threads = 16 float4 each
  {
    const float4* Wv = (const float4*)Wfij;
    const int t = threadIdx.x;
#pragma unroll
    for (int it = 0; it < (DIM * DIM / 4) / 256; ++it) {
      const int   idx = t + it * 256;
      const float4 w  = Wv[idx];
      v4h h;
      h[0] = (_Float16)w.x; h[1] = (_Float16)w.y;
      h[2] = (_Float16)w.z; h[3] = (_Float16)w.w;
      *(v4h*)(sW + idx * 4) = h;                         // ds_store_b64
    }
  }
  __syncthreads();

  const int wave = (int)((blockIdx.x * blockDim.x + threadIdx.x) >> 5);
  if (wave >= etiles) return;                   // wave-uniform (after barrier)
  const int lane = threadIdx.x & 31;
  const int row0 = wave << 4;
  const int n = lane & 15, hi = lane >> 4;

  // A fragments for all 4 K-chunks of this 16-edge tile (fp32 -> f16)
  v16h afrag[4];
#pragma unroll
  for (int kc = 0; kc < 4; ++kc) afrag[kc] = load_a_frag(ein, row0, kc * 32, lane);

  // src/dst indices for the 8 edge rows this lane's accumulator covers
  int sidx[8], didx[8];
#pragma unroll
  for (int r = 0; r < 8; ++r) {
    const int e = row0 + r + 8 * hi;
    sidx[r] = src[e];
    didx[r] = dst[e];
  }

  float sp[8];
#pragma unroll
  for (int r = 0; r < 8; ++r) sp[r] = 0.0f;

#pragma unroll
  for (int nt = 0; nt < 8; ++nt) {
    const int j0 = nt << 4;
    v8f acc = {};
#pragma unroll
    for (int kc = 0; kc < 4; ++kc)
      acc = wmma_f16(afrag[kc], load_b_frag_lds(sW, j0, kc * 32, lane), acc);

    const int   col = j0 + n;
    const float av  = attn[col];
#pragma unroll
    for (int r = 0; r < 8; ++r) {
      float v = acc[r] + xWni[(size_t)sidx[r] * DIM + col]
                       + xWnj[(size_t)didx[r] * DIM + col];
      v = (v > 0.0f) ? v : 0.2f * v;            // leaky_relu(0.2)
      if (fout) fout[(size_t)(row0 + r + 8 * hi) * DIM + col] = v;
      sp[r] += v * av;
    }
  }

  // reduce each row's score over the 16 lanes that hold its columns
#pragma unroll
  for (int m = 1; m < 16; m <<= 1) {
#pragma unroll
    for (int r = 0; r < 8; ++r) sp[r] += __shfl_xor(sp[r], m, 32);
  }
  if (n == 0) {                                 // lanes 0 and 16 write 8 rows each
#pragma unroll
    for (int r = 0; r < 8; ++r) {
      const int e = row0 + r + 8 * hi;
      score[e] = sp[r];
      atomicMax(&menc[didx[r]], enc_f32(sp[r]));
    }
  }
}

// ---------------------------------------------------------------------------
// Kernel 3: a = exp(score - max[dst]); denom[dst] += a  (in-place on score)
// ---------------------------------------------------------------------------
__global__ __launch_bounds__(256)
void egat_edge_exp(float* __restrict__ score, const int* __restrict__ dst,
                   const unsigned* __restrict__ menc,
                   float* __restrict__ denom, int nE) {
  const int e = blockIdx.x * blockDim.x + threadIdx.x;
  if (e >= nE) return;
  const int d = dst[e];
  const float av = __expf(score[e] - dec_f32(menc[d]));
  score[e] = av;
  unsafeAtomicAdd(&denom[d], av);               // global_atomic_add_f32
}

// ---------------------------------------------------------------------------
// Kernel 4: hout[dst] += hproj[src] * (a / denom[dst]).  Wave per edge,
// float4 per lane (32 lanes x 4 = 128), native f32 atomics.
// ---------------------------------------------------------------------------
__global__ __launch_bounds__(256)
void egat_scatter(const float* __restrict__ a, const float* __restrict__ denom,
                  const int* __restrict__ src, const int* __restrict__ dst,
                  const float* __restrict__ hproj, float* __restrict__ hout,
                  int nE) {
  const int e = (int)((blockIdx.x * blockDim.x + threadIdx.x) >> 5);
  if (e >= nE) return;
  const int lane = threadIdx.x & 31;
  const int s = src[e], d = dst[e];
  const float w = a[e] / denom[d];
  const float4 v = ((const float4*)(hproj + (size_t)s * DIM))[lane];
  float* ho = hout + (size_t)d * DIM + lane * 4;
  unsafeAtomicAdd(ho + 0, v.x * w);
  unsafeAtomicAdd(ho + 1, v.y * w);
  unsafeAtomicAdd(ho + 2, v.z * w);
  unsafeAtomicAdd(ho + 3, v.w * w);
}

// ---------------------------------------------------------------------------
// Elementwise helpers
// ---------------------------------------------------------------------------
__global__ __launch_bounds__(256)
void egat_init(unsigned* __restrict__ menc, float* __restrict__ denom,
               float* __restrict__ hout, int nN, int nND) {
  const int i = blockIdx.x * blockDim.x + threadIdx.x;
  if (i < nND) hout[i] = 0.0f;
  if (i < nN) { menc[i] = 0u; denom[i] = 0.0f; }
}

__global__ __launch_bounds__(256)
void egat_relu_add(float* __restrict__ out, const float* __restrict__ a,
                   const float* __restrict__ b, int nelem) {
  const int i = blockIdx.x * blockDim.x + threadIdx.x;
  if (i >= nelem) return;
  const float v = a[i];
  out[i] = (v > 0.0f ? v : 0.0f) + b[i];
}

__global__ __launch_bounds__(256)
void egat_add(float* __restrict__ out, const float* __restrict__ a,
              const float* __restrict__ b, int nelem) {
  const int i = blockIdx.x * blockDim.x + threadIdx.x;
  if (i >= nelem) return;
  out[i] = a[i] + b[i];
}

// ---------------------------------------------------------------------------
// Launcher
// ---------------------------------------------------------------------------
extern "C" void kernel_launch(void* const* d_in, const int* in_sizes, int n_in,
                              void* d_out, int out_size, void* d_ws,
                              size_t ws_size, hipStream_t stream) {
  (void)in_sizes; (void)n_in; (void)out_size; (void)ws_size;

  const float* x     = (const float*)d_in[0];
  const float* efeat = (const float*)d_in[1];
  const int*   src   = (const int*)d_in[2];
  const int*   dst   = (const int*)d_in[3];
  const float* W1ni  = (const float*)d_in[4];
  const float* W1nj  = (const float*)d_in[5];
  const float* W1fij = (const float*)d_in[6];
  const float* W1nd  = (const float*)d_in[7];
  const float* attn1 = (const float*)d_in[8];
  const float* W2ni  = (const float*)d_in[9];
  const float* W2nj  = (const float*)d_in[10];
  const float* W2fij = (const float*)d_in[11];
  const float* W2nd  = (const float*)d_in[12];
  const float* attn2 = (const float*)d_in[13];
  float* out = (float*)d_out;

  const size_t ND = (size_t)NNODES * DIM;      // 6.4M floats
  const size_t ED = (size_t)NEDGES * DIM;      // 102.4M floats

  // workspace layout (floats): fbuf[E*D] | xWni,xWnj,hproj,hout,h0 [N*D each]
  //                            | score[E] | menc[N] | denom[N]
  float* ws     = (float*)d_ws;
  size_t o      = 0;
  float* fbuf   = ws + o; o += ED;
  float* xWni   = ws + o; o += ND;
  float* xWnj   = ws + o; o += ND;
  float* hproj  = ws + o; o += ND;
  float* hout   = ws + o; o += ND;
  float* h0     = ws + o; o += ND;
  float* score  = ws + o; o += (size_t)NEDGES;
  unsigned* menc = (unsigned*)(ws + o); o += (size_t)NNODES;
  float* denom  = ws + o; o += (size_t)NNODES;

  const int ntTiles  = (NNODES / 16) * (DIM / 16);   // 25000 wave-tiles
  const int etTiles  = NEDGES / 16;                  // 50000 wave-tiles
  const int gGemm3   = (ntTiles + 7) / 8;            // 8 waves / block
  const int gEdge    = (etTiles + 7) / 8;            // exactly 6250 full blocks
  const int gScat    = (NEDGES + 7) / 8;             // wave per edge
  const int gE       = (NEDGES + 255) / 256;
  const int gND      = (int)((ND + 255) / 256);
  const int gED      = (int)((ED + 255) / 256);

  // ----------------- Layer 1 -----------------
  egat_init<<<gND, 256, 0, stream>>>(menc, denom, hout, NNODES, (int)ND);
  egat_gemm3<<<gGemm3, 256, 0, stream>>>(x, W1ni, W1nj, W1nd,
                                         xWni, xWnj, hproj, ntTiles);
  egat_edge_fused<<<gEdge, 256, 0, stream>>>(efeat, W1fij, xWni, xWnj, attn1,
                                             src, dst, fbuf, score, menc,
                                             etTiles);
  egat_edge_exp<<<gE, 256, 0, stream>>>(score, dst, menc, denom, NEDGES);
  egat_scatter<<<gScat, 256, 0, stream>>>(score, denom, src, dst, hproj, hout,
                                          NEDGES);
  egat_relu_add<<<gND, 256, 0, stream>>>(h0, hout, x, (int)ND);       // h0
  egat_relu_add<<<gED, 256, 0, stream>>>(fbuf, fbuf, efeat, (int)ED); // h1 in place

  // ----------------- Layer 2 -----------------
  egat_init<<<gND, 256, 0, stream>>>(menc, denom, hout, NNODES, (int)ND);
  egat_gemm3<<<gGemm3, 256, 0, stream>>>(h0, W2ni, W2nj, W2nd,
                                         xWni, xWnj, hproj, ntTiles);
  egat_edge_fused<<<gEdge, 256, 0, stream>>>(fbuf, W2fij, xWni, xWnj, attn2,
                                             src, dst, nullptr, score, menc,
                                             etTiles);
  egat_edge_exp<<<gE, 256, 0, stream>>>(score, dst, menc, denom, NEDGES);
  egat_scatter<<<gScat, 256, 0, stream>>>(score, denom, src, dst, hproj, hout,
                                          NEDGES);
  egat_add<<<gND, 256, 0, stream>>>(out, hout, x, (int)ND);           // + residual
}
